// dGRUCell_58909771432469
// MI455X (gfx1250) — matrix operand order repaired and checked by reading, not compiled, not run on hardware
//
#include <hip/hip_runtime.h>
#include <hip/hip_bf16.h>

// dGRU-like cell: B=131072, DIN=128, DH=32.
// bf16 WMMA (v_wmma_f32_16x16x32_bf16), fp32 accumulate.
// Chained linears are FUSED on-device once per block (W1@W2, b1@W2+b2) using
// WMMA, so the steady-state loop runs only 3 GEMM pairs per 32-row tile:
//   logits = hx@Fwhx + x1@Fwx1 + blog   (N=128, softmax, *x0)
//   gatesd = wdiff@Fdiff + hx@Fhx + bgd (N=96)
//   gatesh = x1@Fx1 + dy@Fdy + bgh      (N=64)
// Fused weights stored transposed ([dout][din], bf16) in LDS: A and B
// fragments are contiguous -> ds_load_b128. 8 waves/block, M=32 per wave.
// Next tile's activations are prefetched into L2 (global_prefetch) while the
// current tile computes. LDS: 90K fused weights + 1.2K bias + 160K scratch.

typedef __bf16 bf16;
typedef __attribute__((ext_vector_type(16))) __bf16 v16bf;
typedef __attribute__((ext_vector_type(8)))  __bf16 v8bf;
typedef __attribute__((ext_vector_type(4)))  __bf16 v4bf;
typedef __attribute__((ext_vector_type(8)))  float  v8f;
typedef __attribute__((ext_vector_type(4)))  float  v4f;

#define BATCH   131072
#define DIN     128
#define DH      32
#define WAVES   8
#define NBLOCKS 256

// fused weight element offsets in LDS (transposed [d3][d1], ld = d1):
#define F_WHX   0       // 128x32  (hx    -> logits)
#define F_WX1   4096    // 128x128 (x1    -> logits)
#define F_DIFF  20480   // 96x128  (wdiff -> gates_d)
#define F_HX    32768   // 96x32   (hx    -> gates_d)
#define F_X1    35840   // 64x128  (x1    -> gates_h)
#define F_DY    44032   // 64x32   (dy    -> gates_h)
#define F_TOTAL 46080
// fused bias offsets (f32): blog[128], bgd[96], bgh[64]
#define BO_LOG  0
#define BO_GD   128
#define BO_GH   224
#define BO_TOT  288

// LDS byte map
#define SM_FW    0                         // bf16 fused weights: 92160 B
#define SM_BIAS  (F_TOTAL * 2)             // 92160: f32 bias 1152 B
#define SM_SCR   (SM_BIAS + BO_TOT * 4)    // 93312: per-wave scratch (16B aligned)
#define SCR_PER_WAVE 10240                 // bf16 elems: X1 4096 | S1 4096 | HX 1024 | DY 1024
#define SM_TOTAL (SM_SCR + WAVES * SCR_PER_WAVE * 2)   // 257152 B

struct KArgs {
    const float* __restrict__ x1;
    const float* __restrict__ x0;
    const float* __restrict__ dx;
    const float* __restrict__ hx;
    const float* __restrict__ w[12];
    const float* __restrict__ b[12];
    float* __restrict__ out;
};

#define FENCE() __builtin_amdgcn_sched_barrier(0)

// ---- fragment loaders (contiguous k = 16*(lane/16) + s in both A and B, wave32) ----

__device__ __forceinline__ v16bf cat16(v8bf lo, v8bf hi) {
    return __builtin_shufflevector(lo, hi, 0, 1, 2, 3, 4, 5, 6, 7,
                                   8, 9, 10, 11, 12, 13, 14, 15);
}

// A: 16xK bf16 row-major in LDS (lda = K stride). Two contiguous 16B runs.
__device__ __forceinline__ v16bf load_a(const bf16* A, int lda, int kb, int lane) {
    const int row = lane & 15;
    const int kh  = (lane >> 4) & 1;
    const bf16* p = A + row * lda + kb + kh * 16;
    v8bf lo = *(const v8bf*)p;
    v8bf hi = *(const v8bf*)(p + 8);
    return cat16(lo, hi);
}

// B from TRANSPOSED weights: BT[n*ldk + k]; 16 contiguous bf16 per lane.
__device__ __forceinline__ v16bf load_bT(const bf16* BT, int ldk, int kb, int nb, int lane) {
    const int col = nb + (lane & 15);
    const int k0  = kb + ((lane >> 4) << 4);
    const bf16* p = BT + col * ldk + k0;
    v8bf lo = *(const v8bf*)p;
    v8bf hi = *(const v8bf*)(p + 8);
    return cat16(lo, hi);
}

// C[m*NT+n] += A(m-th 16xK tile) * B; B fragment reused across MT row-tiles.
template <int MT, int NT, int KT>
__device__ __forceinline__ void gemm(v8f* C, const bf16* A, int lda,
                                     const bf16* BT, int ldk, int lane) {
#pragma unroll
    for (int k = 0; k < KT; ++k) {
        v16bf af[MT];
#pragma unroll
        for (int m = 0; m < MT; ++m) af[m] = load_a(A + m * 16 * lda, lda, k * 32, lane);
#pragma unroll
        for (int n = 0; n < NT; ++n) {
            v16bf bfrag = load_bT(BT, ldk, k * 32, n * 16, lane);
#pragma unroll
            for (int m = 0; m < MT; ++m)
                C[m * NT + n] = __builtin_amdgcn_wmma_f32_16x16x32_bf16(
                    false, af[m], false, bfrag, (short)0, C[m * NT + n], false, false);
        }
    }
}

template <int MT, int NT>
__device__ __forceinline__ void addbias(v8f* C, const float* bias, int lane) {
    const int cl = lane & 15;
#pragma unroll
    for (int n = 0; n < NT; ++n) {
        float bv = bias[n * 16 + cl];
#pragma unroll
        for (int m = 0; m < MT; ++m)
#pragma unroll
            for (int v = 0; v < 8; ++v) C[m * NT + n][v] += bv;
    }
}

__device__ __forceinline__ float sigm(float x) { return 1.0f / (1.0f + __expf(-x)); }

// stage f32 global -> bf16 LDS, vectorized: float4 load, packed bf16x4 store
__device__ __forceinline__ void stage_bf16(bf16* dst, const float* src, int nelem, int lane) {
    const v4f* s4 = (const v4f*)src;
#pragma unroll 4
    for (int i = lane; i < nelem / 4; i += 32) {
        v4f f = s4[i];
        v4bf pk = {(bf16)f.x, (bf16)f.y, (bf16)f.z, (bf16)f.w};
        ((v4bf*)dst)[i] = pk;
    }
}

// prefetch `lines` 128B cachelines starting at p into L2 (1 byte/line/lane)
__device__ __forceinline__ void prefetch_lines(const char* p, int lines, int lane) {
#pragma unroll
    for (int i = lane; i < lines; i += 32)
        __builtin_prefetch(p + (size_t)i * 128, 0, 1);
}

// one fused product F = W1(d1 x d2) @ W2(d2 x d3); W1 row-major bf16 in tmp1,
// W2 transposed bf16 in tmp2; output transposed bf16 FT[n*d1 + m].
// C-layout store is 8 consecutive rows per lane -> one ds_store_b128.
__device__ __forceinline__ void fuse_product(const bf16* W1, const bf16* W2T,
                                             bf16* FT, int d1, int d2, int d3,
                                             int waveid, int lane) {
    const int cl = lane & 15;
    const int rb = (lane >> 4) << 3;
    const int mt = d1 >> 4, nt = d3 >> 4, kt = d2 >> 5;
    for (int t = waveid; t < mt * nt; t += WAVES) {
        const int mi = t % mt, ni = t / mt;
        v8f C = {};
        for (int kb = 0; kb < kt; ++kb) {
            v16bf af = load_a(W1 + mi * 16 * d2, d2, kb * 32, lane);
            v16bf bf_ = load_bT(W2T, d2, kb * 32, ni * 16, lane);
            C = __builtin_amdgcn_wmma_f32_16x16x32_bf16(
                false, af, false, bf_, (short)0, C, false, false);
        }
        bf16* dst = FT + (ni * 16 + cl) * d1 + mi * 16 + rb;
        v8bf pk = {(bf16)C[0], (bf16)C[1], (bf16)C[2], (bf16)C[3],
                   (bf16)C[4], (bf16)C[5], (bf16)C[6], (bf16)C[7]};
        *(v8bf*)dst = pk;
    }
}

__global__ __launch_bounds__(WAVES * 32)
void dgru_cell_kernel(KArgs a) {
    __shared__ __attribute__((aligned(16))) char smem[SM_TOTAL];
    bf16*  sF  = (bf16*)(smem + SM_FW);
    float* sBs = (float*)(smem + SM_BIAS);
    bf16*  scr = (bf16*)(smem + SM_SCR);

    const int tid    = threadIdx.x;
    const int waveid = tid >> 5;
    const int lane   = tid & 31;

    // ---- one-time per block: build fused weights/biases with WMMA ----
    {
        // product table: indices into a.w/a.b per _LAYERS order:
        // 0 diff_0, 1 diff, 2 hx_0, 3 hx, 4 whx_0, 5 whx,
        // 6 wx1_0, 7 wx1, 8 x1_0, 9 x1, 10 dy_0, 11 dy
        const int i1[6]  = {4, 6, 0, 2, 8, 10};
        const int i2[6]  = {5, 7, 1, 3, 9, 11};
        const int pd1[6] = {32, 128, 128, 32, 128, 32};
        const int pd2[6] = {32, 128, 128, 32, 128, 32};
        const int pd3[6] = {128, 128, 96, 96, 64, 64};
        const int pfo[6] = {F_WHX, F_WX1, F_DIFF, F_HX, F_X1, F_DY};
        const int pbo[6] = {BO_LOG, BO_LOG, BO_GD, BO_GD, BO_GH, BO_GH};

        for (int i = tid; i < BO_TOT; i += WAVES * 32) sBs[i] = 0.f;
        bf16* tmp1 = scr;            // up to 16384 elems (32 KB)
        bf16* tmp2 = scr + 16384;    // up to 16384 elems (32 KB)

        for (int p = 0; p < 6; ++p) {
            const int d1 = pd1[p], d2 = pd2[p], d3 = pd3[p];
            const float* w1 = a.w[i1[p]];
            const float* w2 = a.w[i2[p]];
            __syncthreads();
            // stage W1 row-major, W2 transposed (coalesced reads, LDS scatter)
            for (int i = tid; i < d1 * d2; i += WAVES * 32) tmp1[i] = (bf16)w1[i];
            for (int i = tid; i < d2 * d3; i += WAVES * 32) {
                const int k = i / d3, n = i - k * d3;
                tmp2[n * d2 + k] = (bf16)w2[i];
            }
            __syncthreads();
            fuse_product(tmp1, tmp2, sF + pfo[p], d1, d2, d3, waveid, lane);
            // fused bias: bias[n] += b2[n] + sum_k b1[k] * W2[k][n]
            const float* b1 = a.b[i1[p]];
            const float* b2 = a.b[i2[p]];
            for (int n = tid; n < d3; n += WAVES * 32) {
                float acc = b2[n];
                const bf16* col = tmp2 + n * d2;
                for (int k = 0; k < d2; ++k) acc += b1[k] * (float)col[k];
                sBs[pbo[p] + n] += acc;
            }
        }
        __syncthreads();
    }

    const int cl = lane & 15;
    const int rb = (lane >> 4) << 3;

    bf16* X1 = scr + waveid * SCR_PER_WAVE;          // 32x128
    bf16* S1 = X1 + 4096;                            // 32x128 (wdiff)
    bf16* HX = X1 + 8192;                            // 32x32
    bf16* DY = X1 + 9216;                            // 32x32

    const int ntile  = BATCH / 32;                   // M=32 per wave-iteration
    const int stride = gridDim.x * WAVES;
    for (int tile = blockIdx.x * WAVES + waveid; tile < ntile; tile += stride) {
        const int r0 = tile * 32;

        // prefetch next iteration's activations into L2 while this tile computes
        const int nxt = tile + stride;
        if (nxt < ntile) {
            const size_t nr0 = (size_t)nxt * 32;
            prefetch_lines((const char*)(a.x1 + nr0 * DIN), 128, lane);  // 16 KB
            prefetch_lines((const char*)(a.x0 + nr0 * DIN), 128, lane);  // 16 KB
            prefetch_lines((const char*)(a.dx + nr0 * DH), 32, lane);    // 4 KB
            prefetch_lines((const char*)(a.hx + nr0 * DH), 32, lane);    // 4 KB
        }

        // stage activations (vectorized f32->bf16)
        stage_bf16(X1, a.x1 + (size_t)r0 * DIN, 32 * DIN, lane);
        stage_bf16(HX, a.hx + (size_t)r0 * DH, 32 * DH, lane);
        FENCE();

        // ---- logits = hx@Fwhx + x1@Fwx1 + blog ----
        v8f Cl[16] = {};
        gemm<2, 8, 1>(Cl, HX, DH, sF + F_WHX, 32, lane);
        gemm<2, 8, 4>(Cl, X1, DIN, sF + F_WX1, 128, lane);
        addbias<2, 8>(Cl, sBs + BO_LOG, lane);
        FENCE();

        // ---- row softmax over 128 cols; w_diff = softmax * x0 -> S1 ----
#pragma unroll
        for (int m = 0; m < 2; ++m) {
            v8f* Cm = Cl + m * 8;
#pragma unroll
            for (int v = 0; v < 8; ++v) {
                float mx = Cm[0][v];
#pragma unroll
                for (int n = 1; n < 8; ++n) mx = fmaxf(mx, Cm[n][v]);
#pragma unroll
                for (int s = 1; s < 16; s <<= 1) mx = fmaxf(mx, __shfl_xor(mx, s, 32));
                float e[8];
                float sum = 0.f;
#pragma unroll
                for (int n = 0; n < 8; ++n) { e[n] = __expf(Cm[n][v] - mx); sum += e[n]; }
#pragma unroll
                for (int s = 1; s < 16; s <<= 1) sum += __shfl_xor(sum, s, 32);
                const float inv = 1.0f / sum;
                const int lrow = m * 16 + rb + v;
                const float* x0r = a.x0 + (size_t)(r0 + lrow) * DIN;
#pragma unroll
                for (int n = 0; n < 8; ++n)
                    S1[lrow * DIN + n * 16 + cl] = (bf16)(e[n] * inv * x0r[n * 16 + cl]);
            }
        }
        FENCE();

        // ---- gates_d = wdiff@Fdiff + hx@Fhx + bgd  (32x96) ----
        v8f Gd[12] = {};
        gemm<2, 6, 4>(Gd, S1, DIN, sF + F_DIFF, 128, lane);
        gemm<2, 6, 1>(Gd, HX, DH, sF + F_HX, 32, lane);
        addbias<2, 6>(Gd, sBs + BO_GD, lane);
        FENCE();

        // ---- dy = dx*(1-r)+r*c ; hy = o*dy ; dy -> out + LDS bf16 ----
        float hyv[2][2][8];
#pragma unroll
        for (int m = 0; m < 2; ++m) {
#pragma unroll
            for (int t = 0; t < 2; ++t) {
#pragma unroll
                for (int v = 0; v < 8; ++v) {
                    float rd = sigm(Gd[m * 6 + t][v]);
                    float cd = tanhf(Gd[m * 6 + 2 + t][v]);
                    float od = sigm(Gd[m * 6 + 4 + t][v]);
                    const int lrow = m * 16 + rb + v;
                    const int row  = r0 + lrow;
                    const int col  = t * 16 + cl;
                    float dxv = a.dx[(size_t)row * DH + col];
                    float d   = dxv * (1.0f - rd) + rd * cd;
                    hyv[m][t][v] = od * d;
                    a.out[(size_t)row * DH + col] = d;           // output 0: dy
                    DY[lrow * DH + col] = (bf16)d;
                }
            }
        }
        FENCE();

        // ---- gates_h = x1@Fx1 + dy@Fdy + bgh  (32x64) ----
        v8f Gh[8] = {};
        gemm<2, 4, 4>(Gh, X1, DIN, sF + F_X1, 128, lane);
        gemm<2, 4, 1>(Gh, DY, DH, sF + F_DY, 32, lane);
        addbias<2, 4>(Gh, sBs + BO_GH, lane);
        FENCE();

        // ---- cz = hy*(1-rh) + rh*tanh(ch) -> out ----
#pragma unroll
        for (int m = 0; m < 2; ++m) {
#pragma unroll
            for (int t = 0; t < 2; ++t) {
#pragma unroll
                for (int v = 0; v < 8; ++v) {
                    float rh = sigm(Gh[m * 4 + t][v]);
                    float ch = tanhf(Gh[m * 4 + 2 + t][v]);
                    float cz = hyv[m][t][v] * (1.0f - rh) + rh * ch;
                    a.out[(size_t)BATCH * DH + (size_t)(r0 + m * 16 + rb + v) * DH
                          + t * 16 + cl] = cz;
                }
            }
        }
        FENCE();
    }
}

extern "C" void kernel_launch(void* const* d_in, const int* in_sizes, int n_in,
                              void* d_out, int out_size, void* d_ws, size_t ws_size,
                              hipStream_t stream) {
    (void)in_sizes; (void)n_in; (void)out_size; (void)d_ws; (void)ws_size;
    KArgs a;
    a.x1 = (const float*)d_in[0];
    a.x0 = (const float*)d_in[1];
    a.dx = (const float*)d_in[2];
    a.hx = (const float*)d_in[3];
    // params dict insertion order: for each layer in _LAYERS order, _w then _b
    for (int m = 0; m < 12; ++m) {
        a.w[m] = (const float*)d_in[4 + 2 * m];
        a.b[m] = (const float*)d_in[5 + 2 * m];
    }
    a.out = (float*)d_out;
    dgru_cell_kernel<<<NBLOCKS, WAVES * 32, 0, stream>>>(a);
}
